// FullAttention_11905649344760
// MI455X (gfx1250) — compile-verified, matrix operand order checked by reading, not run on hardware
//
#include <hip/hip_runtime.h>

typedef __attribute__((ext_vector_type(16))) __bf16 v16bf;
typedef __attribute__((ext_vector_type(4)))  __bf16 v4bf;
typedef __attribute__((ext_vector_type(8)))  float  v8f;
typedef __attribute__((ext_vector_type(4)))  float  v4f;

#define N_  2
#define L_  2048
#define S_  2048
#define H_  16
#define E_  64
#define D_  64
#define BM  128      // query rows per block (8 waves x 16)
#define BN  64       // key rows per S tile
#define LOG2E 1.4426950408889634f

// Inverse of the documented 16-bit A/B fragment layout:
//   lane-half h2 = (k>>3)&1 ; element slot e_idx = (k&7) + 8*(k>>4)

// Fused DPP xor-butterfly max within each 16-lane row:
// one v_max_num_f32_dpp per step (dst = max(dpp16(src0), src1)).
__device__ __forceinline__ float hmax16(float v) {
    float t;
    asm volatile("v_max_num_f32_dpp %0, %1, %2 row_xmask:8 row_mask:0xf bank_mask:0xf"
                 : "=&v"(t) : "v"(v), "v"(v));
    asm volatile("v_max_num_f32_dpp %0, %1, %2 row_xmask:4 row_mask:0xf bank_mask:0xf"
                 : "=&v"(v) : "v"(t), "v"(t));
    asm volatile("v_max_num_f32_dpp %0, %1, %2 row_xmask:2 row_mask:0xf bank_mask:0xf"
                 : "=&v"(t) : "v"(v), "v"(v));
    asm volatile("v_max_num_f32_dpp %0, %1, %2 row_xmask:1 row_mask:0xf bank_mask:0xf"
                 : "=&v"(v) : "v"(t), "v"(t));
    return v;
}

__global__ __launch_bounds__(256)
void fattn_wmma_kernel(const float* __restrict__ Q, const float* __restrict__ K,
                       const float* __restrict__ V, const float* __restrict__ Mask,
                       float* __restrict__ Out)
{
    __shared__ __align__(32) __bf16 Kf[4 * 2 * 32 * 16];   // K frags (8 KB)
    __shared__ __align__(32) __bf16 Vf[2 * 4 * 32 * 16];   // V frags (8 KB)
    __shared__ __align__(32) __bf16 Pf[8 * 2 * 32 * 16];   // P frags per wave (16 KB)

    const int tid  = threadIdx.x;
    const int w    = tid >> 5;
    const int lane = tid & 31;
    const int h2   = lane >> 4;
    const int mrow = lane & 15;
    const int ncol = lane & 15;

    const int bid = blockIdx.x;
    const int lb  = bid & 15;
    const int h   = (bid >> 4) & 15;
    const int n   = bid >> 8;

    const int l_base = lb * BM + w * 16;
    const float temp = 0.125f;           // 1/sqrt(64)

    __bf16* Pw = Pf + w * (2 * 32 * 16);

    // ---------------- Q fragments (registers, whole kernel) ----------------
    v16bf qa[2];
    {
        const float* qrow = Q + (((size_t)n * L_ + (l_base + mrow)) * H_ + h) * E_;
        #pragma unroll
        for (int half = 0; half < 2; ++half) {
            const float* base = qrow + half * 32 + 8 * h2;
            #pragma unroll
            for (int j = 0; j < 8; ++j)  qa[half][j]     = (__bf16)base[j];
            #pragma unroll
            for (int j = 0; j < 8; ++j)  qa[half][8 + j] = (__bf16)base[16 + j];
        }
    }
    // all-ones B fragment: row-sum via the matrix pipe (P . 1)
    v16bf ones;
    #pragma unroll
    for (int j = 0; j < 16; ++j) ones[j] = (__bf16)1.0f;

    v8f o[4] = {v8f{0.f}, v8f{0.f}, v8f{0.f}, v8f{0.f}};
    float run_max[8], run_max2[8], run_sum[8];
    #pragma unroll
    for (int r = 0; r < 8; ++r) {
        run_max[r] = -3.0e38f; run_max2[r] = -3.0e38f; run_sum[r] = 0.0f;
    }

    #pragma unroll 1
    for (int s0 = 0; s0 < S_; s0 += BN) {
        __syncthreads();   // previous iteration done with Kf/Vf

        // ---------- stage K tile directly into fragment layout ----------
        #pragma unroll
        for (int it = 0; it < 4; ++it) {
            int idx = (it * 256 + tid) * 4;
            int s   = idx >> 6;
            int col = idx & 63;
            const float* gk = K + (((size_t)n * S_ + (s0 + s)) * H_ + h) * E_ + col;
            v4f k4 = *(const v4f*)gk;
            if (s0 + BN < S_) __builtin_prefetch(gk + (size_t)BN * H_ * E_, 0, 1);
            int half = col >> 5;
            int k    = col & 31;
            int kh2  = (k >> 3) & 1;
            int slot = (k & 7) + 8 * (k >> 4);
            int lfr  = (s & 15) + 16 * kh2;
            int st   = s >> 4;
            v4bf pk;
            pk[0] = (__bf16)k4[0]; pk[1] = (__bf16)k4[1];
            pk[2] = (__bf16)k4[2]; pk[3] = (__bf16)k4[3];
            *(v4bf*)&Kf[(((st * 2 + half) * 32 + lfr) << 4) + slot] = pk;
        }
        // ---------- stage V tile: 4x4 (s x d) block per thread ----------
        {
            int d0  = (tid & 15) * 4;
            int sl0 = (tid >> 4) * 4;
            const float* gv = V + (((size_t)n * S_ + (s0 + sl0)) * H_ + h) * D_ + d0;
            v4f r0 = *(const v4f*)(gv + 0 * H_ * D_);
            v4f r1 = *(const v4f*)(gv + 1 * H_ * D_);
            v4f r2 = *(const v4f*)(gv + 2 * H_ * D_);
            v4f r3 = *(const v4f*)(gv + 3 * H_ * D_);
            if (s0 + BN < S_) __builtin_prefetch(gv + (size_t)BN * H_ * D_, 0, 1);
            int kk   = sl0 >> 5;
            int k    = sl0 & 31;
            int kh2  = (k >> 3) & 1;
            int slot = (k & 7) + 8 * (k >> 4);
            #pragma unroll
            for (int j = 0; j < 4; ++j) {
                int d   = d0 + j;
                int lfr = (d & 15) + 16 * kh2;
                int nd  = d >> 4;
                v4bf pv;
                pv[0] = (__bf16)r0[j]; pv[1] = (__bf16)r1[j];
                pv[2] = (__bf16)r2[j]; pv[3] = (__bf16)r3[j];
                *(v4bf*)&Vf[(((kk * 4 + nd) * 32 + lfr) << 4) + slot] = pv;
            }
        }
        __syncthreads();

        // ------------- GEMM1: preloaded fragment pairs, 8 WMMAs -------------
        v8f sc[4];
        #pragma unroll
        for (int stp = 0; stp < 2; ++stp) {
            const int sa = 2 * stp, sb = 2 * stp + 1;
            v16bf b00 = *(const v16bf*)&Kf[((sa * 2 + 0) * 32 + lane) << 4];
            v16bf b01 = *(const v16bf*)&Kf[((sa * 2 + 1) * 32 + lane) << 4];
            v16bf b10 = *(const v16bf*)&Kf[((sb * 2 + 0) * 32 + lane) << 4];
            v16bf b11 = *(const v16bf*)&Kf[((sb * 2 + 1) * 32 + lane) << 4];
            v8f z = {0.f};
            v8f ca = __builtin_amdgcn_wmma_f32_16x16x32_bf16(false, qa[0], false, b00, (short)0, z,  false, false);
            v8f cb = __builtin_amdgcn_wmma_f32_16x16x32_bf16(false, qa[0], false, b10, (short)0, z,  false, false);
            ca     = __builtin_amdgcn_wmma_f32_16x16x32_bf16(false, qa[1], false, b01, (short)0, ca, false, false);
            cb     = __builtin_amdgcn_wmma_f32_16x16x32_bf16(false, qa[1], false, b11, (short)0, cb, false, false);
            sc[sa] = ca; sc[sb] = cb;
        }
        // scale + additive mask
        #pragma unroll
        for (int st = 0; st < 4; ++st) {
            const int sg = s0 + st * 16 + ncol;
            #pragma unroll
            for (int r = 0; r < 8; ++r) {
                sc[st][r] = sc[st][r] * temp + Mask[(size_t)(l_base + r + 8 * h2) * S_ + sg];
            }
        }

        // ------------- online max: fused DPP row_xmask butterflies -------------
        float nm2[8], scl8[8];
        #pragma unroll
        for (int r = 0; r < 8; ++r) {
            float m01 = fmaxf(sc[0][r], sc[1][r]);
            float m23 = fmaxf(sc[2][r], sc[3][r]);
            float tmax = hmax16(fmaxf(m01, m23));
            float nmax = fmaxf(run_max[r], tmax);
            run_max[r] = nmax;
            nm2[r] = nmax * LOG2E;
            scl8[r] = __builtin_amdgcn_exp2f(run_max2[r] - nm2[r]);
            run_max2[r] = nm2[r];
        }

        // exp2 + store P directly in A-fragment layout (bf16)
        #pragma unroll
        for (int st = 0; st < 4; ++st) {
            int kk   = st >> 1;
            int slot = (ncol & 7) + 8 * (st & 1);
            int kh2  = ncol >> 3;
            #pragma unroll
            for (int r = 0; r < 8; ++r) {
                float p = __builtin_amdgcn_exp2f(__builtin_fmaf(sc[st][r], LOG2E, -nm2[r]));
                Pw[((kk * 32 + (r + 8 * h2) + 16 * kh2) << 4) + slot] = (__bf16)p;
            }
        }
        // rescale O while P stores land
        #pragma unroll
        for (int r = 0; r < 8; ++r) {
            o[0][r] *= scl8[r]; o[1][r] *= scl8[r];
            o[2][r] *= scl8[r]; o[3][r] *= scl8[r];
        }

        // ---------------- GEMM2: O += P . V ; row-sum = P . 1 ----------------
        v16bf pa0 = *(const v16bf*)&Pw[(0 * 32 + lane) << 4];
        v16bf pa1 = *(const v16bf*)&Pw[(1 * 32 + lane) << 4];
        v8f z = {0.f};
        v8f rs = __builtin_amdgcn_wmma_f32_16x16x32_bf16(false, pa0, false, ones, (short)0, z,  false, false);
        rs     = __builtin_amdgcn_wmma_f32_16x16x32_bf16(false, pa1, false, ones, (short)0, rs, false, false);
        #pragma unroll
        for (int nd = 0; nd < 4; ++nd) {
            v16bf vb0 = *(const v16bf*)&Vf[((0 * 4 + nd) * 32 + lane) << 4];
            v16bf vb1 = *(const v16bf*)&Vf[((1 * 4 + nd) * 32 + lane) << 4];
            o[nd] = __builtin_amdgcn_wmma_f32_16x16x32_bf16(false, pa0, false, vb0, (short)0, o[nd], false, false);
            o[nd] = __builtin_amdgcn_wmma_f32_16x16x32_bf16(false, pa1, false, vb1, (short)0, o[nd], false, false);
        }
        #pragma unroll
        for (int r = 0; r < 8; ++r) {
            run_sum[r] = run_sum[r] * scl8[r] + rs[r];
        }
    }

    // ---------------- epilogue: divide by running sum, store fp32 ----------------
    #pragma unroll
    for (int r = 0; r < 8; ++r) {
        float inv = 1.0f / run_sum[r];
        size_t base = (((size_t)n * L_ + (l_base + r + 8 * h2)) * H_ + h) * D_;
        #pragma unroll
        for (int nd = 0; nd < 4; ++nd) {
            Out[base + nd * 16 + ncol] = o[nd][r] * inv;
        }
    }
}

extern "C" void kernel_launch(void* const* d_in, const int* in_sizes, int n_in,
                              void* d_out, int out_size, void* d_ws, size_t ws_size,
                              hipStream_t stream) {
    const float* Q = (const float*)d_in[0];
    const float* K = (const float*)d_in[1];
    const float* V = (const float*)d_in[2];
    const float* M = (const float*)d_in[3];
    float* O = (float*)d_out;
    dim3 grid(N_ * H_ * (L_ / BM));   // 2*16*16 = 512 blocks
    fattn_wmma_kernel<<<grid, 256, 0, stream>>>(Q, K, V, M, O);
}
